// AttentionalGNN_54795192762710
// MI455X (gfx1250) — matrix operand order repaired and checked by reading, not compiled, not run on hardware
//
#include <hip/hip_runtime.h>
#include <math.h>

typedef __attribute__((ext_vector_type(16))) _Float16 v16h;
typedef __attribute__((ext_vector_type(8)))  float    v8f;

#define TPB 256

// ---------------------------------------------------------------------------
// Generic WMMA GEMM:  C[m,n] = alpha * sum_k A[k,m]*B[k,n] + bias[m]  (opt ReLU)
// A,B are f32 in memory with arbitrary *32-bit* element strides; converted to
// f16 in registers; accumulation in f32 via v_wmma_f32_16x16x32_f16.
// One wave computes a 16(M) x 64(N) tile: 4 accumulators, A fragment reused
// across 4 B fragments -> 4 WMMAs per 32-deep K step.
// All offset arithmetic is unsigned 32-bit (every buffer here is < 2^26 B),
// so addresses lower to SGPR-base + 32-bit VGPR offset (1 VGPR per address)
// instead of 64-bit VGPR pairs -- this is what keeps the loop free of spills.
// Out-of-range M/N lanes use clamped (in-bounds) addresses; their garbage
// results are never stored. Only the K guard affects valid outputs: the main
// loop over K&~31 is fully unconditional, the tail (K%32!=0) masks via a
// multiply with a 0/1 float so loads cannot be sunk under EXEC branches.
// Fragment layouts per CDNA5 ISA 7.12.2 (wave32):
//   A (16x32 f16): lane L -> M = L%16; half=L/16;
//       elem e -> K = (e&7) + (half<<3) + ((e>>3)<<4)
//   B (32x16 f16): lane L -> N = L%16; elem e -> K = (e&7) + (half<<3) + ((e>>3)<<4)
//   C (16x16 f32): lane L -> N = L%16; elem r -> M = r + (L/16)*8
// ---------------------------------------------------------------------------
__global__ void __launch_bounds__(32)
gemm_atb_wmma(const float* __restrict__ A, int sAk, int sAm,
              const float* __restrict__ B, int sBk, int sBn,
              float* __restrict__ C, int sCm, int sCn,
              const float* __restrict__ bias,
              int M, int N, int K, float alpha, int relu)
{
    const int tn = blockIdx.x * 64;
    const int tm = blockIdx.y * 16;
    const int lane = threadIdx.x & 31;
    const int half = lane >> 4;
    const int l16  = lane & 15;

    v8f acc0 = {}, acc1 = {}, acc2 = {}, acc3 = {};

    const int      am    = tm + l16;
    const unsigned uAk   = (unsigned)sAk;
    const unsigned uBk   = (unsigned)sBk;
    const unsigned amoff = (unsigned)(am < M ? am : 0) * (unsigned)sAm;

    const int  bn0 = tn + l16,      bn1 = tn + 16 + l16;
    const int  bn2 = tn + 32 + l16, bn3 = tn + 48 + l16;
    const bool bv0 = bn0 < N, bv1 = bn1 < N, bv2 = bn2 < N, bv3 = bn3 < N;
    const unsigned bo0 = (unsigned)(bv0 ? bn0 : 0) * (unsigned)sBn;
    const unsigned bo1 = (unsigned)(bv1 ? bn1 : 0) * (unsigned)sBn;
    const unsigned bo2 = (unsigned)(bv2 ? bn2 : 0) * (unsigned)sBn;
    const unsigned bo3 = (unsigned)(bv3 ? bn3 : 0) * (unsigned)sBn;

    const int Ka = K & ~31;

    // ---- main loop: no masking at all ----
    for (int k0 = 0; k0 < Ka; k0 += 32) {
        v16h af;
#pragma unroll
        for (int e = 0; e < 16; ++e) {
            unsigned ka = (unsigned)(k0 + ((e & 7) | (half << 3) | ((e >> 3) << 4)));
            af[e] = (_Float16)A[ka * uAk + amoff];
        }
        v16h bf0, bf1, bf2, bf3;
#pragma unroll
        for (int e = 0; e < 16; ++e) {
            unsigned koff = (unsigned)(k0 + ((e & 7) | (half << 3) | ((e >> 3) << 4))) * uBk;
            bf0[e] = (_Float16)B[koff + bo0];
            bf1[e] = (_Float16)B[koff + bo1];
            bf2[e] = (_Float16)B[koff + bo2];
            bf3[e] = (_Float16)B[koff + bo3];
        }
        acc0 = __builtin_amdgcn_wmma_f32_16x16x32_f16(false, af, false, bf0, (short)0, acc0, false, false);
        acc1 = __builtin_amdgcn_wmma_f32_16x16x32_f16(false, af, false, bf1, (short)0, acc1, false, false);
        acc2 = __builtin_amdgcn_wmma_f32_16x16x32_f16(false, af, false, bf2, (short)0, acc2, false, false);
        acc3 = __builtin_amdgcn_wmma_f32_16x16x32_f16(false, af, false, bf3, (short)0, acc3, false, false);
    }

    // ---- tail: K%32 != 0; mask K overrun by multiplying with 0/1 ----
    if (Ka < K) {
        v16h af;
#pragma unroll
        for (int e = 0; e < 16; ++e) {
            int ka = Ka + ((e & 7) | (half << 3) | ((e >> 3) << 4));
            float ksel = (ka < K) ? 1.0f : 0.0f;
            float va = A[(unsigned)(ka < K ? ka : 0) * uAk + amoff];
            af[e] = (_Float16)(va * ksel);
        }
        v16h bf0, bf1, bf2, bf3;
#pragma unroll
        for (int e = 0; e < 16; ++e) {
            int      kb   = Ka + ((e & 7) | (half << 3) | ((e >> 3) << 4));
            float    ksel = (kb < K) ? 1.0f : 0.0f;
            unsigned koff = (unsigned)(kb < K ? kb : 0) * uBk;
            bf0[e] = (_Float16)(B[koff + bo0] * ksel);
            bf1[e] = (_Float16)(B[koff + bo1] * ksel);
            bf2[e] = (_Float16)(B[koff + bo2] * ksel);
            bf3[e] = (_Float16)(B[koff + bo3] * ksel);
        }
        acc0 = __builtin_amdgcn_wmma_f32_16x16x32_f16(false, af, false, bf0, (short)0, acc0, false, false);
        acc1 = __builtin_amdgcn_wmma_f32_16x16x32_f16(false, af, false, bf1, (short)0, acc1, false, false);
        acc2 = __builtin_amdgcn_wmma_f32_16x16x32_f16(false, af, false, bf2, (short)0, acc2, false, false);
        acc3 = __builtin_amdgcn_wmma_f32_16x16x32_f16(false, af, false, bf3, (short)0, acc3, false, false);
    }

#pragma unroll
    for (int r = 0; r < 8; ++r) {
        int mm = tm + (half << 3) + r;
        if (mm < M) {
            float    bb = bias ? bias[mm] : 0.0f;
            unsigned cm = (unsigned)mm * (unsigned)sCm;
            float v;
            if (bv0) { v = acc0[r] * alpha + bb; if (relu) v = fmaxf(v, 0.0f); C[cm + (unsigned)bn0 * (unsigned)sCn] = v; }
            if (bv1) { v = acc1[r] * alpha + bb; if (relu) v = fmaxf(v, 0.0f); C[cm + (unsigned)bn1 * (unsigned)sCn] = v; }
            if (bv2) { v = acc2[r] * alpha + bb; if (relu) v = fmaxf(v, 0.0f); C[cm + (unsigned)bn2 * (unsigned)sCn] = v; }
            if (bv3) { v = acc3[r] * alpha + bb; if (relu) v = fmaxf(v, 0.0f); C[cm + (unsigned)bn3 * (unsigned)sCn] = v; }
        }
    }
}

// y[p] = bias[0] + sum_c w[c] * x[c*ldx + p]      (conv Cin->1)
__global__ void gemv_rows(const float* __restrict__ w, const float* __restrict__ bias,
                          const float* __restrict__ x, float* __restrict__ y,
                          int K, int ldx, long N)
{
    long p = blockIdx.x * (long)blockDim.x + threadIdx.x;
    if (p >= N) return;
    float acc = bias[0];
    unsigned up = (unsigned)p;
#pragma unroll 8
    for (int c = 0; c < K; ++c) acc += w[c] * x[(unsigned)c * (unsigned)ldx + up];
    y[p] = acc;
}

// in-place row softmax; one block per row
__global__ void softmax_rows(float* __restrict__ x, int cols)
{
    long base = (long)blockIdx.x * cols;
    __shared__ float red[TPB];
    int tid = threadIdx.x;
    float mx = -3.402823466e38f;
    for (int c = tid; c < cols; c += blockDim.x) mx = fmaxf(mx, x[base + c]);
    red[tid] = mx; __syncthreads();
    for (int off = TPB / 2; off; off >>= 1) {
        if (tid < off) red[tid] = fmaxf(red[tid], red[tid + off]);
        __syncthreads();
    }
    mx = red[0]; __syncthreads();
    float sum = 0.0f;
    for (int c = tid; c < cols; c += blockDim.x) {
        float e = __expf(x[base + c] - mx);
        x[base + c] = e; sum += e;
    }
    red[tid] = sum; __syncthreads();
    for (int off = TPB / 2; off; off >>= 1) {
        if (tid < off) red[tid] += red[tid + off];
        __syncthreads();
    }
    float inv = 1.0f / red[0];
    for (int c = tid; c < cols; c += blockDim.x) x[base + c] *= inv;
}

// replicate reference dist1 = swapaxes(dist,1,2).reshape(b,-1,n0*n1):
// out[r*32768 + c] = dist[d, i, j],  i=r>>4, dh=r&15, dl=c>>12, j=c&4095, d=dh*8+dl
__global__ void dist1_gather(const float* __restrict__ dist, float* __restrict__ out)
{
    long t = blockIdx.x * (long)blockDim.x + threadIdx.x;
    if (t >= 128L * 32768L) return;
    int  r = (int)(t >> 15);
    long c = t & 32767;
    int i  = r >> 4;
    int dh = r & 15;
    int dl = (int)(c >> 12);
    int j  = (int)(c & 4095);
    int d  = dh * 8 + dl;
    out[t] = dist[((long)d * 8 + i) * 4096 + j];
}

// cat[c*P + p]: c<128 -> q[c*nq + p/nk]; c<256 -> kv[(c-128)*nk + p%nk]; else dg[(c-256)*P + p]
__global__ void cross_concat(const float* __restrict__ q, int nq,
                             const float* __restrict__ kv, int nk,
                             const float* __restrict__ dg,
                             float* __restrict__ out, long P)
{
    long t = blockIdx.x * (long)blockDim.x + threadIdx.x;
    if (t >= 384L * P) return;
    int  c = (int)(t / P);
    long p = t - (long)c * P;
    float v;
    if (c < 128)      v = q[(long)c * nq + (p / nk)];
    else if (c < 256) v = kv[(long)(c - 128) * nk + (p % nk)];
    else              v = dg[(long)(c - 256) * P + p];
    out[t] = v;
}

__global__ void copyk(float* __restrict__ dst, const float* __restrict__ src, long n)
{
    long t = blockIdx.x * (long)blockDim.x + threadIdx.x;
    if (t < n) dst[t] = src[t];
}

__global__ void zerok(float* __restrict__ dst, long n)
{
    long t = blockIdx.x * (long)blockDim.x + threadIdx.x;
    if (t < n) dst[t] = 0.0f;
}

// x += a (+ b)
__global__ void addk(float* __restrict__ x, const float* __restrict__ a,
                     const float* __restrict__ b, long n)
{
    long t = blockIdx.x * (long)blockDim.x + threadIdx.x;
    if (t < n) x[t] += a[t] + (b ? b[t] : 0.0f);
}

// log_softmax over the row dimension (axis=-2): one block per column
__global__ void logsoftmax_cols(const float* __restrict__ s, float* __restrict__ out,
                                int rows, int cols)
{
    int j = blockIdx.x;
    __shared__ float red[TPB];
    int tid = threadIdx.x;
    float mx = -3.402823466e38f;
    for (int i = tid; i < rows; i += blockDim.x) mx = fmaxf(mx, s[(long)i * cols + j]);
    red[tid] = mx; __syncthreads();
    for (int off = TPB / 2; off; off >>= 1) {
        if (tid < off) red[tid] = fmaxf(red[tid], red[tid + off]);
        __syncthreads();
    }
    mx = red[0]; __syncthreads();
    float sum = 0.0f;
    for (int i = tid; i < rows; i += blockDim.x) sum += __expf(s[(long)i * cols + j] - mx);
    red[tid] = sum; __syncthreads();
    for (int off = TPB / 2; off; off >>= 1) {
        if (tid < off) red[tid] += red[tid + off];
        __syncthreads();
    }
    float lse = mx + logf(red[0]);
    for (int i = tid; i < rows; i += blockDim.x)
        out[(long)i * cols + j] = s[(long)i * cols + j] - lse;
}

// couplings (m+1)x(n+1): inner block = ls, border = alpha
__global__ void build_couplings(const float* __restrict__ ls, const float* __restrict__ alpha,
                                float* __restrict__ c, int m, int n)
{
    long t = blockIdx.x * (long)blockDim.x + threadIdx.x;
    long total = (long)(m + 1) * (n + 1);
    if (t >= total) return;
    int i = (int)(t / (n + 1));
    int j = (int)(t - (long)i * (n + 1));
    c[t] = (i < m && j < n) ? ls[(long)i * n + j] : alpha[0];
}

// u[i] = log_mu(i) - logsumexp_j(c[i,j] + v[j]); small n -> serial per row
__global__ void ot_u(const float* __restrict__ c, const float* __restrict__ v,
                     float* __restrict__ u, int m, int n, float norm, float lmu_last)
{
    int i = blockIdx.x * blockDim.x + threadIdx.x;
    if (i > m) return;
    const float* row = c + (long)i * (n + 1);
    float mx = -3.402823466e38f;
    for (int j = 0; j <= n; ++j) mx = fmaxf(mx, row[j] + v[j]);
    float s = 0.0f;
    for (int j = 0; j <= n; ++j) s += __expf(row[j] + v[j] - mx);
    float lmu = (i < m) ? norm : lmu_last;
    u[i] = lmu - (mx + logf(s));
}

// v[j] = log_nu(j) - logsumexp_i(c[i,j] + u[i]); one block per column
__global__ void ot_v(const float* __restrict__ c, const float* __restrict__ u,
                     float* __restrict__ v, int m, int n, float norm, float lnu_last)
{
    int j = blockIdx.x;
    __shared__ float red[TPB];
    int tid = threadIdx.x;
    float mx = -3.402823466e38f;
    for (int i = tid; i <= m; i += blockDim.x)
        mx = fmaxf(mx, c[(long)i * (n + 1) + j] + u[i]);
    red[tid] = mx; __syncthreads();
    for (int off = TPB / 2; off; off >>= 1) {
        if (tid < off) red[tid] = fmaxf(red[tid], red[tid + off]);
        __syncthreads();
    }
    mx = red[0]; __syncthreads();
    float s = 0.0f;
    for (int i = tid; i <= m; i += blockDim.x)
        s += __expf(c[(long)i * (n + 1) + j] + u[i] - mx);
    red[tid] = s; __syncthreads();
    for (int off = TPB / 2; off; off >>= 1) {
        if (tid < off) red[tid] += red[tid + off];
        __syncthreads();
    }
    if (tid == 0) {
        float lnu = (j < n) ? norm : lnu_last;
        v[j] = lnu - (mx + logf(red[0]));
    }
}

__global__ void build_z(const float* __restrict__ c, const float* __restrict__ u,
                        const float* __restrict__ v, float* __restrict__ z,
                        int m, int n, float norm)
{
    long t = blockIdx.x * (long)blockDim.x + threadIdx.x;
    if (t >= (long)m * n) return;
    int i = (int)(t / n);
    int j = (int)(t - (long)i * n);
    z[t] = c[(long)i * (n + 1) + j] + u[i] + v[j] - norm;
}

__global__ void minmax_kernel(const float* __restrict__ x, long n, float* __restrict__ red2)
{
    __shared__ float smn[TPB], smx[TPB];
    int tid = threadIdx.x;
    float mn = 3.402823466e38f, mx = -3.402823466e38f;
    for (long t = tid; t < n; t += blockDim.x) {
        float v = x[t];
        mn = fminf(mn, v);
        mx = fmaxf(mx, v);
    }
    smn[tid] = mn; smx[tid] = mx; __syncthreads();
    for (int off = TPB / 2; off; off >>= 1) {
        if (tid < off) {
            smn[tid] = fminf(smn[tid], smn[tid + off]);
            smx[tid] = fmaxf(smx[tid], smx[tid + off]);
        }
        __syncthreads();
    }
    if (tid == 0) { red2[0] = smn[0]; red2[1] = smx[0]; }
}

__global__ void final_out(const float* __restrict__ z, const float* __restrict__ red2,
                          const int* __restrict__ mask, float* __restrict__ out, int n)
{
    int j = threadIdx.x;
    if (j >= n) return;
    float mf = mask[j] ? 0.0f : 1.0f;
    out[j] = (z[j] + (red2[0] - red2[1] - 40.0f) * mf) * 15.0f;
}

// ---------------------------------------------------------------------------
// Host-side orchestration
// ---------------------------------------------------------------------------
namespace {

constexpr int cN0 = 8, cN1 = 4096, cN2 = 256, cN3 = 256;
constexpr long cP01 = (long)cN0 * cN1; // 32768

struct SelfP {
    const float *mw, *mb, *qw, *qb, *kw, *kb, *vw, *vb, *m0w, *m0b, *m1w, *m1b;
};
struct CrossP {
    const float *a0w, *a0b, *a1w, *a1b, *m0w, *m0b, *m1w, *m1b;
};

struct Bufs {
    float *D0, *D1, *D2, *D3;
    float *dist1g;
    float *BIG;        // 16.78M floats: MHA per-head scores OR cross concat (aliased)
    float *hid128;     // 128 x 32768
    float *bq, *bk, *bv, *bo;
    float *msg;
    float *cat2, *hid2;
    float *scbuf;      // 32768: cross-attention raw scores
    float *score1;     // last cross layer desc1-direction scores (4096 x 8)
    float *del0, *del1, *del2, *del3, *del12, *del03, *del21, *del30;
    float *coup, *u, *v, *zbuf, *red2;
};

struct LeafCtx {
    void* const* d_in;
    int li;
    long off;
    const float* packed; // non-null => params passed as one blob at d_in[5]
};

static const float* nextLeaf(LeafCtx& c, long nelem)
{
    const float* p = c.packed ? (c.packed + c.off)
                              : (const float*)c.d_in[5 + c.li];
    c.li++;
    c.off += nelem;
    return p;
}

static void parseSelf(LeafCtx& c, SelfP& p)
{
    p.mw  = nextLeaf(c, 128 * 128); p.mb  = nextLeaf(c, 128);
    p.qw  = nextLeaf(c, 128 * 128); p.qb  = nextLeaf(c, 128);
    p.kw  = nextLeaf(c, 128 * 128); p.kb  = nextLeaf(c, 128);
    p.vw  = nextLeaf(c, 128 * 128); p.vb  = nextLeaf(c, 128);
    p.m0w = nextLeaf(c, 256 * 256); p.m0b = nextLeaf(c, 256);
    p.m1w = nextLeaf(c, 128 * 256); p.m1b = nextLeaf(c, 128);
}

static void parseCross(LeafCtx& c, CrossP& p)
{
    p.a0w = nextLeaf(c, 128 * 384); p.a0b = nextLeaf(c, 128);
    p.a1w = nextLeaf(c, 128);       p.a1b = nextLeaf(c, 1);
    p.m0w = nextLeaf(c, 256 * 256); p.m0b = nextLeaf(c, 256);
    p.m1w = nextLeaf(c, 128 * 256); p.m1b = nextLeaf(c, 128);
}

static inline dim3 g1(long n) { return dim3((unsigned)((n + TPB - 1) / TPB)); }

static void gemm(hipStream_t st,
                 const float* A, int sAk, int sAm,
                 const float* B, int sBk, int sBn,
                 float* C, int sCm, int sCn,
                 const float* bias, int M, int N, int K, float alpha, int relu)
{
    dim3 grid((N + 63) / 64, (M + 15) / 16);
    gemm_atb_wmma<<<grid, dim3(32), 0, st>>>(A, sAk, sAm, B, sBk, sBn,
                                             C, sCm, sCn, bias, M, N, K, alpha, relu);
}

// pointwise conv: Y(MxN) = W(MxK) * X(KxN) + b   (X,Y channel-major)
static void conv(hipStream_t st, const float* W, const float* b,
                 const float* X, float* Y, int M, int K, int N, bool relu)
{
    gemm(st, W, 1, K, X, N, 1, Y, N, 1, b, M, N, K, 1.0f, relu ? 1 : 0);
}

// multi-head attention: msg(128 x nq) from q(128 x nq), kv(128 x nk)
static void mha(hipStream_t st, const SelfP& p, const float* q, int nq,
                const float* kv, int nk, Bufs& B)
{
    conv(st, p.qw, p.qb, q,  B.bq, 128, 128, nq, false);
    conv(st, p.kw, p.kb, kv, B.bk, 128, 128, nk, false);
    conv(st, p.vw, p.vb, kv, B.bv, 128, 128, nk, false);
    const float scale = 0.17677669529663688f; // 1/sqrt(32)
    for (int h = 0; h < 4; ++h) {
        // scores_h[n,m] = scale * sum_d Q[(4d+h),n]*K[(4d+h),m]
        gemm(st, B.bq + (long)h * nq, 4 * nq, 1,
                 B.bk + (long)h * nk, 4 * nk, 1,
                 B.BIG, nk, 1, nullptr, nq, nk, 32, scale, 0);
        softmax_rows<<<nq, TPB, 0, st>>>(B.BIG, nk);
        // Out[(4d+h),n] = sum_m V[(4d+h),m] * P[n,m]
        gemm(st, B.bv + (long)h * nk, 1, 4 * nk,
                 B.BIG, 1, nk,
                 B.bo + (long)h * nq, 4 * nq, 1, nullptr, 32, nq, nk, 1.0f, 0);
    }
    conv(st, p.mw, p.mb, B.bo, B.msg, 128, 128, nq, false);
}

// cross MLP-attention: msg(128 x nq) from q(128 x nq), kv(128 x nk), dist feature dg(128 x nq*nk)
static void cross_attn(hipStream_t st, const CrossP& p, const float* q, int nq,
                       const float* kv, int nk, const float* dg,
                       float* score_keep, Bufs& B)
{
    long P = (long)nq * nk;
    cross_concat<<<g1(384L * P), TPB, 0, st>>>(q, nq, kv, nk, dg, B.BIG, P);
    conv(st, p.a0w, p.a0b, B.BIG, B.hid128, 128, 384, (int)P, true);
    gemv_rows<<<g1(P), TPB, 0, st>>>(p.a1w, p.a1b, B.hid128, B.scbuf, 128, (int)P, P);
    if (score_keep)
        copyk<<<g1(P), TPB, 0, st>>>(score_keep, B.scbuf, P);
    softmax_rows<<<nq, TPB, 0, st>>>(B.scbuf, nk);
    // msg[d,n] = sum_m prob[n,m] * kv[d,m]
    gemm(st, kv, 1, nk,
             B.scbuf, 1, nk,
             B.msg, nq, 1, nullptr, 128, nq, nk, 1.0f, 0);
}

// delta = MLP_256_256_128( concat[x ; msg] )
static void prop_mlp(hipStream_t st, const float* m0w, const float* m0b,
                     const float* m1w, const float* m1b,
                     const float* x, float* delta, int N, Bufs& B)
{
    copyk<<<g1(128L * N), TPB, 0, st>>>(B.cat2, x, 128L * N);
    copyk<<<g1(128L * N), TPB, 0, st>>>(B.cat2 + 128L * N, B.msg, 128L * N);
    conv(st, m0w, m0b, B.cat2, B.hid2, 256, 256, N, true);
    conv(st, m1w, m1b, B.hid2, delta, 128, 256, N, false);
}

static void prop_self(hipStream_t st, const SelfP& p, const float* x, int nq,
                      const float* src, int nk, float* delta, Bufs& B)
{
    mha(st, p, x, nq, src, nk, B);
    prop_mlp(st, p.m0w, p.m0b, p.m1w, p.m1b, x, delta, nq, B);
}

static void prop_cross(hipStream_t st, const CrossP& p, const float* x, int nq,
                       const float* src, int nk, const float* dg,
                       float* delta, float* score_keep, Bufs& B)
{
    cross_attn(st, p, x, nq, src, nk, dg, score_keep, B);
    prop_mlp(st, p.m0w, p.m0b, p.m1w, p.m1b, x, delta, nq, B);
}

} // namespace

extern "C" void kernel_launch(void* const* d_in, const int* in_sizes, int n_in,
                              void* d_out, int out_size, void* d_ws, size_t ws_size,
                              hipStream_t stream)
{
    (void)in_sizes; (void)out_size; (void)ws_size;

    const float* desc0 = (const float*)d_in[0];
    const float* desc1 = (const float*)d_in[1];
    const float* desc2 = (const float*)d_in[2];
    const float* desc3 = (const float*)d_in[3];
    const float* dist  = (const float*)d_in[4];

    // ---- parameter parsing (pytree flatten: attn, bin_score, ghattn, phattn) ----
    LeafCtx ctx;
    ctx.d_in = d_in;
    ctx.li = 0;
    ctx.off = 0;
    ctx.packed = (n_in < 20) ? (const float*)d_in[5] : nullptr;

    SelfP apS[6]; CrossP apC[6]; SelfP gh[6], ph[6];
    for (int i = 0; i < 6; ++i) {
        if ((i & 1) == 0) parseSelf(ctx, apS[i]);
        else              parseCross(ctx, apC[i]);
    }
    const float* alpha = nextLeaf(ctx, 1); // bin_score
    for (int i = 0; i < 6; ++i) parseSelf(ctx, gh[i]);
    for (int i = 0; i < 6; ++i) parseSelf(ctx, ph[i]);
    const int* mask = (const int*)(ctx.packed ? d_in[6] : d_in[5 + ctx.li]);

    // ---- workspace carving ----
    float* W = (float*)d_ws;
    long o = 0;
    auto take = [&](long n) { float* p = W + o; o += n; return p; };

    Bufs B;
    B.D0 = take(128L * cN0);  B.D1 = take(128L * cN1);
    B.D2 = take(128L * cN2);  B.D3 = take(128L * cN3);
    B.dist1g = take(128L * cP01);
    B.BIG    = take(16777216L);      // max(4096x4096 scores, 384x32768 concat)
    B.hid128 = take(128L * cP01);
    B.bq = take(128L * cN1); B.bk = take(128L * cN1);
    B.bv = take(128L * cN1); B.bo = take(128L * cN1);
    B.msg  = take(128L * cN1);
    B.cat2 = take(256L * cN1); B.hid2 = take(256L * cN1);
    B.scbuf  = take(cP01);
    B.score1 = take(cP01);
    B.del0  = take(128L * cN0); B.del1  = take(128L * cN1);
    B.del2  = take(128L * cN2); B.del3  = take(128L * cN3);
    B.del12 = take(128L * cN1); B.del03 = take(128L * cN0);
    B.del21 = take(128L * cN2); B.del30 = take(128L * cN3);
    B.coup = take(4097L * 9); B.u = take(4097); B.v = take(9);
    B.zbuf = take(cP01); B.red2 = take(2);

    // ---- init: copy descriptors, build dist1 gather ----
    copyk<<<g1(128L * cN0), TPB, 0, stream>>>(B.D0, desc0, 128L * cN0);
    copyk<<<g1(128L * cN1), TPB, 0, stream>>>(B.D1, desc1, 128L * cN1);
    copyk<<<g1(128L * cN2), TPB, 0, stream>>>(B.D2, desc2, 128L * cN2);
    copyk<<<g1(128L * cN3), TPB, 0, stream>>>(B.D3, desc3, 128L * cN3);
    dist1_gather<<<g1(128L * cP01), TPB, 0, stream>>>(dist, B.dist1g);

    // ---- 6 GNN layers (self, cross, self, cross, self, cross) ----
    for (int L = 0; L < 6; ++L) {
        if ((L & 1) == 0) { // self
            prop_self(stream, apS[L], B.D0, cN0, B.D0, cN0, B.del0, B);
            prop_self(stream, apS[L], B.D1, cN1, B.D1, cN1, B.del1, B);
            prop_self(stream, ph[L],  B.D2, cN2, B.D2, cN2, B.del2, B);
            prop_self(stream, gh[L],  B.D3, cN3, B.D3, cN3, B.del3, B);
            addk<<<g1(128L * cN0), TPB, 0, stream>>>(B.D0, B.del0, nullptr, 128L * cN0);
            addk<<<g1(128L * cN1), TPB, 0, stream>>>(B.D1, B.del1, nullptr, 128L * cN1);
            addk<<<g1(128L * cN2), TPB, 0, stream>>>(B.D2, B.del2, nullptr, 128L * cN2);
            addk<<<g1(128L * cN3), TPB, 0, stream>>>(B.D3, B.del3, nullptr, 128L * cN3);
        } else { // cross
            prop_cross(stream, apC[L], B.D0, cN0, B.D1, cN1, dist,     B.del0, nullptr,  B);
            prop_cross(stream, apC[L], B.D1, cN1, B.D0, cN0, B.dist1g, B.del1, B.score1, B);
            prop_self(stream, ph[L], B.D2, cN2, B.D1, cN1, B.del21, B);
            prop_self(stream, ph[L], B.D1, cN1, B.D2, cN2, B.del12, B);
            prop_self(stream, gh[L], B.D0, cN0, B.D3, cN3, B.del03, B);
            prop_self(stream, gh[L], B.D3, cN3, B.D0, cN0, B.del30, B);
            addk<<<g1(128L * cN2), TPB, 0, stream>>>(B.D2, B.del21, nullptr, 128L * cN2);
            addk<<<g1(128L * cN3), TPB, 0, stream>>>(B.D3, B.del30, nullptr, 128L * cN3);
            addk<<<g1(128L * cN0), TPB, 0, stream>>>(B.D0, B.del0, B.del03, 128L * cN0);
            addk<<<g1(128L * cN1), TPB, 0, stream>>>(B.D1, B.del1, B.del12, 128L * cN1);
        }
    }

    // ---- optimal transport (log-domain Sinkhorn, 5 iters) + readout ----
    const int m = cN1, n = cN0; // score1 is (4096 x 8)
    const float norm = -logf((float)(m + n));
    const float lmu_last = logf((float)n) + norm;
    const float lnu_last = logf((float)m) + norm;

    logsoftmax_cols<<<n, TPB, 0, stream>>>(B.score1, B.scbuf, m, n);
    build_couplings<<<g1((long)(m + 1) * (n + 1)), TPB, 0, stream>>>(B.scbuf, alpha, B.coup, m, n);
    zerok<<<g1(m + 1), TPB, 0, stream>>>(B.u, m + 1);
    zerok<<<g1(n + 1), TPB, 0, stream>>>(B.v, n + 1);
    for (int it = 0; it < 5; ++it) {
        ot_u<<<g1(m + 1), TPB, 0, stream>>>(B.coup, B.v, B.u, m, n, norm, lmu_last);
        ot_v<<<n + 1, TPB, 0, stream>>>(B.coup, B.u, B.v, m, n, norm, lnu_last);
    }
    build_z<<<g1((long)m * n), TPB, 0, stream>>>(B.coup, B.u, B.v, B.zbuf, m, n, norm);
    minmax_kernel<<<1, TPB, 0, stream>>>(B.zbuf, (long)m * n, B.red2);
    final_out<<<1, 32, 0, stream>>>(B.zbuf, B.red2, mask, (float*)d_out, n);
}
